// WindowAttention_78829829751136
// MI455X (gfx1250) — matrix operand order repaired and checked by reading, not compiled, not run on hardware
//
#include <hip/hip_runtime.h>
#include <hip/hip_bf16.h>

typedef __attribute__((ext_vector_type(16))) __bf16 v16bf;
typedef __attribute__((ext_vector_type(8)))  __bf16 v8bf;
typedef __attribute__((ext_vector_type(8)))  float  v8f;

#define WHS   7
#define N_TOK 49
#define C_DIM 256
#define H_NUM 8
#define HD    32
#define NW    1024

// LDS strides in bf16 elements (padded for bank-conflict-free 16B loads)
#define XSTR 264   // 256 + 8
#define QSTR 40    // 32 + 8
#define VSTR 72    // 64 + 8
#define PSTR 72    // 64 + 8

// ---- WMMA fragment loaders -------------------------------------------------
// A operand (16x32 bf16, row-major [row][K] source): lane l holds row (l&15);
// elements 0..7 -> K = khi*8 + e, elements 8..15 -> K = 16 + khi*8 + (e-8).
__device__ __forceinline__ v16bf load_fragA(const __bf16* base, int stride, int lane) {
  int r = lane & 15, khi = lane >> 4;
  const v8bf* p0 = (const v8bf*)(base + r * stride + khi * 8);
  const v8bf* p1 = (const v8bf*)(base + r * stride + khi * 8 + 16);
  union { v16bf w; v8bf h[2]; } u;
  u.h[0] = *p0; u.h[1] = *p1;
  return u.w;
}
// B operand (32x16 bf16, source stored row-major as [N][K] i.e. B^T):
// lane l holds column n=(l&15); elements 0..7 -> K = khi*8 + e, 8..15 -> K = 16 + khi*8 + (e-8).
__device__ __forceinline__ v16bf load_fragB(const __bf16* base, int stride, int lane) {
  int n = lane & 15, khi = lane >> 4;
  const v8bf* p0 = (const v8bf*)(base + n * stride + khi * 8);
  const v8bf* p1 = (const v8bf*)(base + n * stride + khi * 8 + 16);
  union { v16bf w; v8bf h[2]; } u;
  u.h[0] = *p0; u.h[1] = *p1;
  return u.w;
}

__device__ __forceinline__ v8f wmma_bf16(v16bf a, v16bf b, v8f c) {
  return __builtin_amdgcn_wmma_f32_16x16x32_bf16(false, a, false, b, (short)0, c, false, false);
}

// ---- Pre-pass: pack W^T into per-lane A-fragment order ---------------------
// out[((ct*8+kt)*32+lane)*16 + e] = bf16( W[k][c] )   (W row-major [K][C])
//   c = ct*16 + (lane&15)
//   k = kt*32 + (e&8 ? 16 : 0) + (lane>>4)*8 + (e&7)     (A K-pattern)
__global__ void pack_w_fragsA(const float* __restrict__ W, __bf16* __restrict__ out,
                              int wcols, int total) {
  int idx = blockIdx.x * blockDim.x + threadIdx.x;
  if (idx >= total) return;
  int e    = idx & 15;
  int lane = (idx >> 4) & 31;
  int kt   = (idx >> 9) & 7;
  int ct   = idx >> 12;
  int c = ct * 16 + (lane & 15);
  int k = kt * 32 + ((e & 8) ? 16 : 0) + ((lane >> 4) << 3) + (e & 7);
  out[idx] = (__bf16)W[k * wcols + c];
}

// ---- Pre-pass: relative position bias [H][49][49] --------------------------
// Reference: idx = 13*((jp-jq)+(ip-iq)+12), JAX gather clamps OOB -> min(.,168)
__global__ void build_bias(const float* __restrict__ table, float* __restrict__ bias) {
  int idx = blockIdx.x * blockDim.x + threadIdx.x;
  if (idx >= H_NUM * N_TOK * N_TOK) return;
  int h   = idx / (N_TOK * N_TOK);
  int rem = idx % (N_TOK * N_TOK);
  int p = rem / N_TOK, q = rem % N_TOK;
  int ip = p / WHS, jp = p % WHS, iq = q / WHS, jq = q % WHS;
  int r = 13 * ((jp - jq) + (ip - iq) + 12);
  if (r > 168) r = 168;
  bias[idx] = table[r * H_NUM + h];
}

// ---- Fused window-attention kernel: 1 block = 1 window ---------------------
__global__ __launch_bounds__(256, 1)
void win_attn_fused(const float* __restrict__ x, const int* __restrict__ mask,
                    const float* __restrict__ qkv_b, const float* __restrict__ proj_b,
                    const __bf16* __restrict__ wqkvA, const __bf16* __restrict__ wprojA,
                    const float* __restrict__ biasA, float* __restrict__ out) {
  extern __shared__ __align__(16) __bf16 smem[];
  __bf16* Xb = smem;                       // 64*XSTR     (reused as O after attention)
  __bf16* Qs = Xb + 64 * XSTR;             // 8*64*QSTR   Q, pre-scaled, bf16
  __bf16* Ks = Qs + H_NUM * 64 * QSTR;     // 8*64*QSTR
  __bf16* Vt = Ks + H_NUM * 64 * QSTR;     // 8*32*VSTR   V transposed [h][hd][tok]
  __bf16* Ps = Vt + H_NUM * HD * VSTR;     // 8*16*PSTR   per-wave softmax block

  const int tid  = threadIdx.x;
  const int wave = tid >> 5;
  const int lane = tid & 31;
  const int nlo  = lane & 15;
  const int khi  = lane >> 4;
  const int win  = blockIdx.x;
  const int wmsk = win & (NW - 1);

  // Stage A: x window -> bf16 LDS, rows 49..63 zeroed
  {
    const float4* xw = (const float4*)(x + (size_t)win * N_TOK * C_DIM);
    for (int v = tid; v < 64 * (C_DIM / 4); v += 256) {
      int row = v >> 6;
      int c4  = (v & 63) << 2;
      float4 f = make_float4(0.f, 0.f, 0.f, 0.f);
      if (row < N_TOK) f = xw[row * (C_DIM / 4) + (v & 63)];
      __bf16* d = Xb + row * XSTR + c4;
      d[0] = (__bf16)f.x; d[1] = (__bf16)f.y; d[2] = (__bf16)f.z; d[3] = (__bf16)f.w;
    }
  }
  __syncthreads();

  // Stage B (transposed): QKV^T tile [16 chan][16 tok] = Wqkv^T(A) x X(B).
  // Bias pre-loaded INTO the accumulator (C-layout rows = output channels).
  // unroll 2 bounds the weight-prefetch window (2x64 regs) -> no scratch spills.
  for (int mt = 0; mt < 4; ++mt) {
    v16bf bx[8];
#pragma unroll
    for (int kt = 0; kt < 8; ++kt)
      bx[kt] = load_fragB(Xb + mt * 16 * XSTR + kt * 32, XSTR, lane);
    int tok = mt * 16 + nlo;
#pragma unroll 2
    for (int j = 0; j < 6; ++j) {
      int ct = wave * 6 + j;
      int c0 = ct * 16 + 8 * khi;
      float4 b0 = *(const float4*)(qkv_b + c0);
      float4 b1 = *(const float4*)(qkv_b + c0 + 4);
      v8f acc = {b0.x, b0.y, b0.z, b0.w, b1.x, b1.y, b1.z, b1.w};
#pragma unroll
      for (int kt = 0; kt < 8; ++kt) {
        v16bf a = *(const v16bf*)(wqkvA + (((ct * 8 + kt) * 32 + lane) << 4));
        acc = wmma_bf16(a, bx[kt], acc);
      }
      // Tile-uniform metadata, forced to SGPRs -> scalar branches.
      int sec  = __builtin_amdgcn_readfirstlane(ct >> 4);   // 0=q 1=k 2=v
      int head = __builtin_amdgcn_readfirstlane((ct >> 1) & 7);
      int hd0  = ((ct & 1) << 4) + 8 * khi;                 // 8 consecutive hd
      if (sec == 0) {
        v8bf pk;
#pragma unroll
        for (int r = 0; r < 8; ++r)
          pk[r] = (__bf16)(acc[r] * 0.17677669529663687f);
        *(v8bf*)(Qs + (head * 64 + tok) * QSTR + hd0) = pk;
      } else if (sec == 1) {
        v8bf pk;
#pragma unroll
        for (int r = 0; r < 8; ++r)
          pk[r] = (__bf16)acc[r];
        *(v8bf*)(Ks + (head * 64 + tok) * QSTR + hd0) = pk;
      } else {
        __bf16* dst = Vt + (head * HD + hd0) * VSTR + tok;
#pragma unroll
        for (int r = 0; r < 8; ++r)
          dst[r * VSTR] = (__bf16)acc[r];
      }
    }
  }
  __syncthreads();

  // Stage C: attention, one head per wave
  {
    const int h = wave;
    const float* biasH = biasA + h * N_TOK * N_TOK;
    const int*   maskW = mask + wmsk * N_TOK * N_TOK;
    const __bf16* Qh = Qs + h * 64 * QSTR;
    const __bf16* Kh = Ks + h * 64 * QSTR;
    const __bf16* Vh = Vt + h * HD * VSTR;
    __bf16* Ph = Ps + h * 16 * PSTR;
    for (int mt = 0; mt < 4; ++mt) {
      v16bf aq = load_fragA(Qh + mt * 16 * QSTR, QSTR, lane);  // K = hd = 32, one step
      v8f att[4];
#pragma unroll
      for (int nt = 0; nt < 4; ++nt) {
        v16bf bk = load_fragB(Kh + nt * 16 * QSTR, QSTR, lane);
        v8f z = {};
        att[nt] = wmma_bf16(aq, bk, z);
      }
      // bias + mask + row softmax (rows live in 16-lane halves of the wave)
#pragma unroll
      for (int r = 0; r < 8; ++r) {
        int i = mt * 16 + r + 8 * khi;
        float l[4];
#pragma unroll
        for (int nt = 0; nt < 4; ++nt) {
          int j = nt * 16 + nlo;
          float v = att[nt][r];
          if (j >= N_TOK)      v = -1e30f;                 // padded key column
          else if (i >= N_TOK) v = 0.f;                    // padded query row (unused)
          else v += biasH[i * N_TOK + j] - 100.f * (float)maskW[i * N_TOK + j];
          l[nt] = v;
        }
        float rmax = fmaxf(fmaxf(l[0], l[1]), fmaxf(l[2], l[3]));
#pragma unroll
        for (int m = 1; m < 16; m <<= 1) rmax = fmaxf(rmax, __shfl_xor(rmax, m, 32));
        float s = 0.f;
#pragma unroll
        for (int nt = 0; nt < 4; ++nt) { float e = __expf(l[nt] - rmax); l[nt] = e; s += e; }
#pragma unroll
        for (int m = 1; m < 16; m <<= 1) s += __shfl_xor(s, m, 32);
        float rinv = 1.f / s;
        int mrow = r + 8 * khi;
#pragma unroll
        for (int nt = 0; nt < 4; ++nt)
          Ph[mrow * PSTR + nt * 16 + nlo] = (__bf16)(l[nt] * rinv);
      }
      // O block = P @ V  (K = 64 tokens = 2 steps; 2 hd tiles) -> reuse Xb as O
#pragma unroll
      for (int ntv = 0; ntv < 2; ++ntv) {
        v8f oacc = {};
#pragma unroll
        for (int kt = 0; kt < 2; ++kt) {
          v16bf ap = load_fragA(Ph + kt * 32, PSTR, lane);
          v16bf bv = load_fragB(Vh + ntv * 16 * VSTR + kt * 32, VSTR, lane);
          oacc = wmma_bf16(ap, bv, oacc);
        }
#pragma unroll
        for (int r = 0; r < 8; ++r) {
          int tok = mt * 16 + r + 8 * khi;
          Xb[tok * XSTR + h * 32 + ntv * 16 + nlo] = (__bf16)oacc[r];
        }
      }
    }
  }
  __syncthreads();

  // Stage D (transposed): out^T tile [16 chan][16 tok] = Wproj^T(A) x O(B).
  // proj_b pre-loaded into accumulator; two float4 global stores per tile.
  {
    float* outW = out + (size_t)win * N_TOK * C_DIM;
    for (int mt = 0; mt < 4; ++mt) {
      v16bf bx[8];
#pragma unroll
      for (int kt = 0; kt < 8; ++kt)
        bx[kt] = load_fragB(Xb + mt * 16 * XSTR + kt * 32, XSTR, lane);
      int tok = mt * 16 + nlo;
#pragma unroll
      for (int j = 0; j < 2; ++j) {
        int ct = wave * 2 + j;
        int c0 = ct * 16 + 8 * khi;
        float4 p0 = *(const float4*)(proj_b + c0);
        float4 p1 = *(const float4*)(proj_b + c0 + 4);
        v8f acc = {p0.x, p0.y, p0.z, p0.w, p1.x, p1.y, p1.z, p1.w};
#pragma unroll
        for (int kt = 0; kt < 8; ++kt) {
          v16bf a = *(const v16bf*)(wprojA + (((ct * 8 + kt) * 32 + lane) << 4));
          acc = wmma_bf16(a, bx[kt], acc);
        }
        if (tok < N_TOK) {
          *(float4*)(outW + tok * C_DIM + c0) =
              make_float4(acc[0], acc[1], acc[2], acc[3]);
          *(float4*)(outW + tok * C_DIM + c0 + 4) =
              make_float4(acc[4], acc[5], acc[6], acc[7]);
        }
      }
    }
  }
}

extern "C" void kernel_launch(void* const* d_in, const int* in_sizes, int n_in,
                              void* d_out, int out_size, void* d_ws, size_t ws_size,
                              hipStream_t stream) {
  (void)in_sizes; (void)n_in; (void)out_size; (void)ws_size;
  const float* x      = (const float*)d_in[0];
  const int*   mask   = (const int*)d_in[1];
  const float* qkv_w  = (const float*)d_in[2];
  const float* qkv_b  = (const float*)d_in[3];
  const float* proj_w = (const float*)d_in[4];
  const float* proj_b = (const float*)d_in[5];
  const float* btable = (const float*)d_in[6];
  float* out = (float*)d_out;

  char* ws = (char*)d_ws;
  __bf16* wqkvA  = (__bf16*)ws;                      // 48*8*32*16 bf16 = 393216 B
  __bf16* wprojA = (__bf16*)(ws + 393216);           // 16*8*32*16 bf16 = 131072 B
  float*  biasA  = (float*)(ws + 393216 + 131072);   // 8*49*49 f32    =  76832 B

  pack_w_fragsA<<<(196608 + 255) / 256, 256, 0, stream>>>(qkv_w, wqkvA, 768, 196608);
  pack_w_fragsA<<<(65536 + 255) / 256, 256, 0, stream>>>(proj_w, wprojA, 256, 65536);
  build_bias<<<(19208 + 255) / 256, 256, 0, stream>>>(btable, biasA);

  const int smem =
      (64 * XSTR + 2 * H_NUM * 64 * QSTR + H_NUM * HD * VSTR + H_NUM * 16 * PSTR) * 2;
  win_attn_fused<<<8192, 256, smem, stream>>>(x, mask, qkv_b, proj_b,
                                              wqkvA, wprojA, biasA, out);
}
